// GenNetGenotypeEncoder_61263413510612
// MI455X (gfx1250) — matrix operand order repaired and checked by reading, not compiled, not run on hardware
//
#include <hip/hip_runtime.h>
#include <stdint.h>

// ---------------- problem constants ----------------
#define BATCH   256
#define GENO    75000
#define GENES   10000
#define NEDGE   75000
#define HDIM    512
#define FFDIM   2048
#define KP_G    10016          // GENES padded to multiple of 32
#define LN_EPS  1e-5f

// ---------------- WMMA types ----------------
typedef __bf16 v16bf __attribute__((ext_vector_type(16)));
typedef float  v8f   __attribute__((ext_vector_type(8)));
typedef unsigned int uint32x4 __attribute__((ext_vector_type(4)));

union BF16x16 { v16bf v; uint32x4 q[2]; };

__device__ __forceinline__ uint16_t f2bf(float f) {
    uint32_t u = __float_as_uint(f);
    uint32_t r = u + 0x7FFFu + ((u >> 16) & 1u);   // round-to-nearest-even
    return (uint16_t)(r >> 16);
}

// ---------------- 1) transpose + convert weights: src (K x N, f32) -> dst (N x Kpad, bf16) ----------------
__global__ void convert_transpose_bf16(const float* __restrict__ src,
                                       uint16_t* __restrict__ dst,
                                       int K, int N, int Kpad) {
    int idx = blockIdx.x * blockDim.x + threadIdx.x;
    int total = N * Kpad;
    if (idx >= total) return;
    int n = idx / Kpad;
    int k = idx - n * Kpad;
    dst[idx] = (k < K) ? f2bf(src[(size_t)k * N + n]) : (uint16_t)0;
}

// ---------------- 2) sparse gene aggregation: LDS segment-sum via ds_add_f32 ----------------
__global__ void gene_accum(const float* __restrict__ x,
                           const int*   __restrict__ snp_idx,
                           const int*   __restrict__ gene_idx,
                           const float* __restrict__ edge_w,
                           const float* __restrict__ gene_b,
                           uint16_t*    __restrict__ gene_bf) {
    __shared__ float acc[KP_G];                       // 40 KB of the 320 KB WGP LDS
    const int b = blockIdx.x;
    const int tid = threadIdx.x;
    const int nt = blockDim.x;

    for (int i = tid; i < KP_G; i += nt) acc[i] = 0.0f;
    __syncthreads();

    const float* xr = x + (size_t)b * GENO;
    for (int e = tid; e < NEDGE; e += nt) {
        int   s = snp_idx[e];
        int   g = gene_idx[e];
        float v = xr[s] * edge_w[e];
        atomicAdd(&acc[g], v);                        // ds_add_f32
    }
    __syncthreads();

    uint16_t* out = gene_bf + (size_t)b * KP_G;
    for (int k = tid; k < KP_G; k += nt) {
        float r = (k < GENES) ? fmaxf(acc[k] + gene_b[k], 0.0f) : 0.0f;
        out[k] = f2bf(r);
    }
}

// ---------------- 3) WMMA bf16 GEMM, 1 wave = one 16x64 C tile (A fragment reused 4x) ----------------
// A: (M x Kpad) bf16 row-major, Bw: (N x Kpad) bf16 row-major ("transposed" weights)
// epilogue modes:
//   0: v = acc + bias[n]                                                     -> bf16 out
//   1: v = relu((acc + bias[n] - mean[n])*g[n]*rsqrt(var[n]+eps) + beta[n])  -> f32 + bf16
//   2: v = acc + bias[n] + resid[m,n]                                        -> f32 out
//   3: v = relu(acc + bias[n])                                               -> bf16 out
//   4: v = acc + bias[n] + resid[m,n]                                        -> f32 out
__global__ void gemm_bf16_wmma(const uint16_t* __restrict__ A,
                               const uint16_t* __restrict__ Bw,
                               int M, int N, int Kpad,
                               const float* __restrict__ bias,
                               const float* __restrict__ bnm,
                               const float* __restrict__ bnv,
                               const float* __restrict__ bng,
                               const float* __restrict__ bnb,
                               const float* __restrict__ resid,
                               float*    __restrict__ outF,
                               uint16_t* __restrict__ outBf,
                               int mode) {
    const int lane = threadIdx.x & 31;
    const int wave = threadIdx.x >> 5;
    const int tile = blockIdx.x * 4 + wave;
    const int ntn  = N >> 6;                // 64-wide N tiles
    const int tm   = tile / ntn;
    const int tn   = tile - tm * ntn;

    const int mrow = lane & 15;
    const int half = lane >> 4;

    const uint16_t* Arow = A + (size_t)(tm * 16 + mrow) * Kpad;
    const uint16_t* Bbase = Bw + (size_t)(tn * 64 + mrow) * Kpad;
    const size_t bstep = (size_t)16 * Kpad;

    v8f acc[4] = {};
    for (int k0 = 0; k0 < Kpad; k0 += 32) {
        BF16x16 a;
        const uint16_t* ap = Arow + k0 + half * 8;    // 16-bit A 16x32 lane striping
        a.q[0] = *(const uint32x4*)(ap);
        a.q[1] = *(const uint32x4*)(ap + 16);

        BF16x16 b[4];
        #pragma unroll
        for (int j = 0; j < 4; ++j) {
            const uint16_t* bp = Bbase + (size_t)j * bstep + k0 + half * 8;
            b[j].q[0] = *(const uint32x4*)(bp);
            b[j].q[1] = *(const uint32x4*)(bp + 16);
        }
        #pragma unroll
        for (int j = 0; j < 4; ++j) {
            acc[j] = __builtin_amdgcn_wmma_f32_16x16x32_bf16(
                /*neg_a=*/false, a.v, /*neg_b=*/false, b[j].v,
                /*c_mod=*/(short)0, acc[j], /*reuse_a=*/false, /*reuse_b=*/false);
        }
    }

    // C/D layout: lanes 0-15 -> N=lane, M=r; lanes 16-31 -> N=lane-16, M=r+8
    const int mbase = tm * 16 + half * 8;

    #pragma unroll
    for (int j = 0; j < 4; ++j) {
        const int n = tn * 64 + j * 16 + (lane & 15);
        const float bia = bias ? bias[n] : 0.0f;
        float scale = 1.0f, mean = 0.0f, beta = 0.0f;
        if (mode == 1) {
            scale = bng[n] * rsqrtf(bnv[n] + LN_EPS);
            mean  = bnm[n];
            beta  = bnb[n];
        }
        #pragma unroll
        for (int r = 0; r < 8; ++r) {
            const size_t idx = (size_t)(mbase + r) * N + n;
            float v = acc[j][r] + bia;
            if (mode == 1)      { v = fmaxf((v - mean) * scale + beta, 0.0f); }
            else if (mode == 2 || mode == 4) { v += resid[idx]; }
            else if (mode == 3) { v = fmaxf(v, 0.0f); }
            if (outF)  outF[idx]  = v;
            if (outBf) outBf[idx] = f2bf(v);
        }
    }
}

// ---------------- 4) per-row LayerNorm (H=512, one 512-thread block per row) ----------------
__global__ void layernorm_row(const float* __restrict__ s,
                              const float* __restrict__ g,
                              const float* __restrict__ b,
                              float*    __restrict__ outF,
                              uint16_t* __restrict__ outBf) {
    __shared__ float r1[HDIM];
    __shared__ float r2[HDIM];
    const int row = blockIdx.x;
    const int t   = threadIdx.x;
    const float v = s[(size_t)row * HDIM + t];
    r1[t] = v;
    r2[t] = v * v;
    __syncthreads();
    for (int off = HDIM / 2; off > 0; off >>= 1) {
        if (t < off) { r1[t] += r1[t + off]; r2[t] += r2[t + off]; }
        __syncthreads();
    }
    const float mu  = r1[0] * (1.0f / HDIM);
    const float var = r2[0] * (1.0f / HDIM) - mu * mu;
    const float y   = (v - mu) * rsqrtf(var + LN_EPS) * g[t] + b[t];
    outF[(size_t)row * HDIM + t] = y;
    if (outBf) outBf[(size_t)row * HDIM + t] = f2bf(y);
}

// ---------------- launch ----------------
extern "C" void kernel_launch(void* const* d_in, const int* in_sizes, int n_in,
                              void* d_out, int out_size, void* d_ws, size_t ws_size,
                              hipStream_t stream) {
    const float* x    = (const float*)d_in[0];
    const int*   snp  = (const int*)  d_in[1];
    const int*   gidx = (const int*)  d_in[2];
    const float* ew   = (const float*)d_in[3];
    const float* gb   = (const float*)d_in[4];
    const float* w1   = (const float*)d_in[5];
    const float* b1   = (const float*)d_in[6];
    const float* bng  = (const float*)d_in[7];
    const float* bnb  = (const float*)d_in[8];
    const float* bnm  = (const float*)d_in[9];
    const float* bnv  = (const float*)d_in[10];
    // wq/bq/wk/bk (11..14) are mathematically dead: softmax over a length-1 axis is identically 1.
    const float* wv   = (const float*)d_in[15];
    const float* bv   = (const float*)d_in[16];
    const float* wo   = (const float*)d_in[17];
    const float* bo   = (const float*)d_in[18];
    const float* ln1g = (const float*)d_in[19];
    const float* ln1b = (const float*)d_in[20];
    const float* ln2g = (const float*)d_in[21];
    const float* ln2b = (const float*)d_in[22];
    const float* ffw1 = (const float*)d_in[23];
    const float* ffb1 = (const float*)d_in[24];
    const float* ffw2 = (const float*)d_in[25];
    const float* ffb2 = (const float*)d_in[26];

    // ---- workspace carve (256B aligned) ----
    char* p = (char*)d_ws;
    auto carve = [&](size_t bytes) -> char* {
        char* r = p;
        p += (bytes + 255) & ~(size_t)255;
        return r;
    };
    uint16_t* w1t    = (uint16_t*)carve((size_t)HDIM  * KP_G  * 2);
    uint16_t* wvt    = (uint16_t*)carve((size_t)HDIM  * HDIM  * 2);
    uint16_t* wot    = (uint16_t*)carve((size_t)HDIM  * HDIM  * 2);
    uint16_t* ffw1t  = (uint16_t*)carve((size_t)FFDIM * HDIM  * 2);
    uint16_t* ffw2t  = (uint16_t*)carve((size_t)HDIM  * FFDIM * 2);
    uint16_t* genebf = (uint16_t*)carve((size_t)BATCH * KP_G  * 2);
    float*    h_f    = (float*)   carve((size_t)BATCH * HDIM  * 4);
    uint16_t* h_bf   = (uint16_t*)carve((size_t)BATCH * HDIM  * 2);
    uint16_t* v_bf   = (uint16_t*)carve((size_t)BATCH * HDIM  * 2);
    float*    s1     = (float*)   carve((size_t)BATCH * HDIM  * 4);
    float*    x1_f   = (float*)   carve((size_t)BATCH * HDIM  * 4);
    uint16_t* x1_bf  = (uint16_t*)carve((size_t)BATCH * HDIM  * 2);
    uint16_t* ff1_bf = (uint16_t*)carve((size_t)BATCH * FFDIM * 2);
    float*    s2     = (float*)   carve((size_t)BATCH * HDIM  * 4);

    const int CT = 256;
    auto cgrid = [&](size_t total) { return (unsigned)((total + CT - 1) / CT); };

    // ---- weight conversion (transpose to N x Kpad bf16) ----
    convert_transpose_bf16<<<cgrid((size_t)HDIM * KP_G),  CT, 0, stream>>>(w1,   w1t,   GENES, HDIM,  KP_G);
    convert_transpose_bf16<<<cgrid((size_t)HDIM * HDIM),  CT, 0, stream>>>(wv,   wvt,   HDIM,  HDIM,  HDIM);
    convert_transpose_bf16<<<cgrid((size_t)HDIM * HDIM),  CT, 0, stream>>>(wo,   wot,   HDIM,  HDIM,  HDIM);
    convert_transpose_bf16<<<cgrid((size_t)FFDIM * HDIM), CT, 0, stream>>>(ffw1, ffw1t, HDIM,  FFDIM, HDIM);
    convert_transpose_bf16<<<cgrid((size_t)HDIM * FFDIM), CT, 0, stream>>>(ffw2, ffw2t, FFDIM, HDIM,  FFDIM);

    // ---- sparse segment-sum into genes (LDS atomics), relu+bias, emit bf16 ----
    gene_accum<<<BATCH, 512, 0, stream>>>(x, snp, gidx, ew, gb, genebf);

    // ---- GEMM chain (4 waves / block, 1 wave = one 16x64 tile) ----
    auto gblocks = [](int M, int N) { return (unsigned)(((M / 16) * (N / 64)) / 4); };

    // h = relu(BN(gene @ w1 + b1))
    gemm_bf16_wmma<<<gblocks(BATCH, HDIM), 128, 0, stream>>>(
        genebf, w1t, BATCH, HDIM, KP_G, b1, bnm, bnv, bng, bnb, nullptr, h_f, h_bf, 1);
    // v = h @ wv + bv         (attention with seqlen 1: ctx == v exactly)
    gemm_bf16_wmma<<<gblocks(BATCH, HDIM), 128, 0, stream>>>(
        h_bf, wvt, BATCH, HDIM, HDIM, bv, nullptr, nullptr, nullptr, nullptr, nullptr, nullptr, v_bf, 0);
    // s1 = h + (v @ wo + bo)
    gemm_bf16_wmma<<<gblocks(BATCH, HDIM), 128, 0, stream>>>(
        v_bf, wot, BATCH, HDIM, HDIM, bo, nullptr, nullptr, nullptr, nullptr, h_f, s1, nullptr, 2);
    // x1 = LN(s1)
    layernorm_row<<<BATCH, HDIM, 0, stream>>>(s1, ln1g, ln1b, x1_f, x1_bf);
    // ff1 = relu(x1 @ ffw1 + ffb1)
    gemm_bf16_wmma<<<gblocks(BATCH, FFDIM), 128, 0, stream>>>(
        x1_bf, ffw1t, BATCH, FFDIM, HDIM, ffb1, nullptr, nullptr, nullptr, nullptr, nullptr, nullptr, ff1_bf, 3);
    // s2 = x1 + (ff1 @ ffw2 + ffb2)
    gemm_bf16_wmma<<<gblocks(BATCH, HDIM), 128, 0, stream>>>(
        ff1_bf, ffw2t, BATCH, HDIM, FFDIM, ffb2, nullptr, nullptr, nullptr, nullptr, x1_f, s2, nullptr, 4);
    // out = LN(s2)
    layernorm_row<<<BATCH, HDIM, 0, stream>>>(s2, ln2g, ln2b, (float*)d_out, nullptr);
}
